// modi_cgcnn_edge_46248207843561
// MI455X (gfx1250) — matrix operand order repaired and checked by reading, not compiled, not run on hardware
//
#include <hip/hip_runtime.h>

// ---------------------------------------------------------------------------
// CGCNN edge block for MI455X (gfx1250): wave32, v_wmma_f32_16x16x32_f16,
// TDM tensor_load_to_lds for tile staging (edge / S tiles), s_wait_tensorcnt,
// W_full B-fragments staged in LDS to keep WMMA deps on short ds latency.
// ---------------------------------------------------------------------------

typedef __attribute__((ext_vector_type(16))) _Float16 v16h;
typedef __attribute__((ext_vector_type(8)))  _Float16 v8h;
typedef __attribute__((ext_vector_type(4)))  _Float16 h4;
typedef __attribute__((ext_vector_type(8)))  float    v8f;
typedef __attribute__((ext_vector_type(4)))  float    f4;
typedef __attribute__((ext_vector_type(4)))  unsigned int u32x4;
typedef __attribute__((ext_vector_type(8)))  int      i32x8;
typedef __attribute__((ext_vector_type(4)))  int      i32x4;

#define WMMA_F16(A, B, C) \
  __builtin_amdgcn_wmma_f32_16x16x32_f16(false, (A), false, (B), (short)0, (C), false, false)

static __device__ __forceinline__ v16h cat8(v8h lo, v8h hi) {
  return __builtin_shufflevector(lo, hi, 0,1,2,3,4,5,6,7,8,9,10,11,12,13,14,15);
}

static __device__ __forceinline__ int lower_bound_dev(const int* a, int n, int key) {
  int lo = 0, hi = n;
  while (lo < hi) { int mid = (lo + hi) >> 1; if (a[mid] < key) lo = mid + 1; else hi = mid; }
  return lo;
}

// ---------------------------------------------------------------------------
// TDM: load a [rows x cols] 2-D tile (row stride == cols) into LDS with
// per-row padding (LDS stride = 2^(pad_int+1) + pad_amt+1 DWORDs).
// OOB rows (>= rows) are zero-filled by the TDM.
// ---------------------------------------------------------------------------
static __device__ __forceinline__ void tdm_load_tile_to_lds(
    const void* gsrc, unsigned lds_off, int rows, int cols,
    int dsize_code, int pad_int_code, int pad_amt_code) {
  unsigned long long ga = (unsigned long long)(size_t)gsrc;
  u32x4 g0;
  g0[0] = 1u;                                             // count=1 (valid)
  g0[1] = lds_off;                                        // lds_addr (bytes)
  g0[2] = (unsigned)ga;                                   // global_addr[31:0]
  g0[3] = (unsigned)((ga >> 32) & 0x01ffffffu) | (2u << 30); // [56:32] | type=2
  i32x8 g1;
  g1[0] = (dsize_code << 16) | (1 << 20) |                // data_size, pad_en
          (pad_int_code << 22) | (pad_amt_code << 25);    // pad interval/amount
  g1[1] = (cols << 16);                                   // tensor_dim0[15:0]
  g1[2] = ((rows & 0xffff) << 16);                        // tensor_dim1[15:0]
  g1[3] = ((rows >> 16) & 0xffff) | (cols << 16);         // td1 hi | tile_dim0
  g1[4] = 128;                                            // tile_dim1=128
  g1[5] = cols;                                           // tensor_dim0_stride
  g1[6] = 0; g1[7] = 0;
  i32x4 z4; z4[0] = z4[1] = z4[2] = z4[3] = 0;
#if defined(__clang_major__) && (__clang_major__ >= 23)
  i32x8 z8; z8[0]=z8[1]=z8[2]=z8[3]=z8[4]=z8[5]=z8[6]=z8[7]=0;
  __builtin_amdgcn_tensor_load_to_lds(g0, g1, z4, z4, z8, 0);
#else
  __builtin_amdgcn_tensor_load_to_lds(g0, g1, z4, z4, 0);
#endif
}

// ---------------------------------------------------------------------------
// K0: weight repack (f32 -> f16 WMMA B-fragments) + zero stat accumulators.
// B-frag (16-bit B, 32x16 KxN): lane = n + 16*hi, half j -> K = (j<8 ? hi*8+j : 16+hi*8+j-8).
// ---------------------------------------------------------------------------
__global__ __launch_bounds__(256) void k_prep(
    const float* __restrict__ Wfull, const float* __restrict__ rW1,
    const float* __restrict__ rW2,
    _Float16* __restrict__ wfull_f, _Float16* __restrict__ w1_f,
    _Float16* __restrict__ w2_f,
    float* __restrict__ sum1, float* __restrict__ sq1,
    float* __restrict__ sum2, float* __restrict__ sq2) {
  int gid = blockIdx.x * blockDim.x + threadIdx.x;
  if (gid < 16384) {  // W_full [128,128] -> [kt:4][nt:8][lane:32][j:16]
    int j = gid & 15, lane = (gid >> 4) & 31, nt = (gid >> 9) & 7, kt = gid >> 12;
    int hi = lane >> 4, nl = lane & 15;
    int k = kt * 32 + hi * 8 + (j & 7) + ((j >> 3) << 4);
    int n = nt * 16 + nl;
    wfull_f[gid] = (_Float16)Wfull[k * 128 + n];
  }
  if (gid < 4096) {  // rW1 [2,64,32] -> [layer:2][kt:2][nt:2][lane][j]
    int j = gid & 15, lane = (gid >> 4) & 31, nt = (gid >> 9) & 1,
        kt = (gid >> 10) & 1, layer = gid >> 11;
    int hi = lane >> 4, nl = lane & 15;
    int k = kt * 32 + hi * 8 + (j & 7) + ((j >> 3) << 4);
    int n = nt * 16 + nl;
    w1_f[gid] = (_Float16)rW1[(layer * 64 + k) * 32 + n];
  }
  if (gid < 4096) {  // rW2 [2,32,64] -> [layer:2][nt:4][lane][j]
    int j = gid & 15, lane = (gid >> 4) & 31, nt = (gid >> 9) & 3, layer = gid >> 11;
    int hi = lane >> 4, nl = lane & 15;
    int k = hi * 8 + (j & 7) + ((j >> 3) << 4);
    int n = nt * 16 + nl;
    w2_f[gid] = (_Float16)rW2[(layer * 32 + k) * 64 + n];
  }
  if (gid < 32768) { sum1[gid] = 0.f; sq1[gid] = 0.f; }
  if (gid < 16384) { sum2[gid] = 0.f; sq2[gid] = 0.f; }
}

// ---------------------------------------------------------------------------
// K1: gated[E,128] = [atom[i1]-atom[i0] | edge] @ W_full  (f16 out).
// edge half of A via TDM (f32 tile, padded stride); W_full fragments staged
// in LDS so every WMMA operand dependency is a short ds_load.
// ---------------------------------------------------------------------------
#define LDA2  72    // halves stride, diff tile (f16)
#define LDE   68    // floats stride, TDM edge tile (64 dw + 4 dw pad)
#define LDO16 136   // halves stride, output tile (aliases edge tile)

__global__ __launch_bounds__(256) void k_gemm1(
    const float* __restrict__ atom, const float* __restrict__ edge,
    const int* __restrict__ nidx, const _Float16* __restrict__ wfrag,
    _Float16* __restrict__ g1, int E) {
  // 18432 (At) + 34816 (Ef / Dt alias) + 32768 (Wl) = 86016 B
  __shared__ __align__(16) char Smem[128 * LDA2 * 2 + 128 * LDE * 4 + 32768];
  _Float16* At = (_Float16*)Smem;                     // 128 x 72 halves
  float*    Ef = (float*)(Smem + 128 * LDA2 * 2);     // 128 x 68 floats
  _Float16* Dt = (_Float16*)(Smem + 128 * LDA2 * 2);  // alias: 128 x 136 halves
  _Float16* Wl = (_Float16*)(Smem + 128 * LDA2 * 2 + 128 * LDE * 4);  // 16384 h
  const int t = threadIdx.x;
  const int base = blockIdx.x * 128;
  int rows = E - base; if (rows > 128) rows = 128;

  if (t < 32) {  // wave 0 kicks the DMA for the edge tile
    tdm_load_tile_to_lds(edge + (size_t)base * 64, (unsigned)(size_t)(void*)Ef,
                         rows, 64, /*4B*/2, /*64dw*/5, /*+4dw*/3);
  }

  {  // stage W_full fragments into LDS (32 KB, coalesced b128)
    const v8h* src = (const v8h*)wfrag;
    v8h* dst = (v8h*)Wl;
#pragma unroll
    for (int i = 0; i < 8; ++i) dst[t + i * 256] = src[t + i * 256];
  }

  {  // stage diff half (f16) while the TDM runs
    int er = t >> 1, ch = t & 1;
    int e = base + er;
    _Float16* dst = &At[er * LDA2 + ch * 32];
    if (e < E) {
      int i0 = nidx[2 * e], i1 = nidx[2 * e + 1];
      const f4* a0 = (const f4*)(atom + (size_t)i0 * 64 + ch * 32);
      const f4* a1 = (const f4*)(atom + (size_t)i1 * 64 + ch * 32);
#pragma unroll
      for (int j = 0; j < 8; ++j) {
        f4 d = a1[j] - a0[j];
        *(h4*)(dst + j * 4) = __builtin_convertvector(d, h4);
      }
    } else {
      h4 z; z[0] = z[1] = z[2] = z[3] = (_Float16)0.f;
#pragma unroll
      for (int j = 0; j < 8; ++j) *(h4*)(dst + j * 4) = z;
    }
  }
  if (t < 32) __builtin_amdgcn_s_wait_tensorcnt((short)0);
  __syncthreads();

  const int wid = t >> 5, lane = t & 31;
  const int ml = lane & 15, hi = lane >> 4;
  const int m0 = wid * 16;

  v8f acc[8];
#pragma unroll
  for (int nt = 0; nt < 8; ++nt)
#pragma unroll
    for (int r = 0; r < 8; ++r) acc[nt][r] = 0.f;

#pragma unroll
  for (int kt = 0; kt < 4; ++kt) {
    v16h a;
    if (kt < 2) {  // diff half from f16 LDS
      const _Float16* ap = &At[(m0 + ml) * LDA2 + kt * 32 + hi * 8];
      a = cat8(*(const v8h*)ap, *(const v8h*)(ap + 16));
    } else {       // edge half from TDM f32 tile, convert on the fly
      const float* ep = &Ef[(m0 + ml) * LDE + (kt - 2) * 32 + hi * 8];
      h4 q0 = __builtin_convertvector(*(const f4*)(ep +  0), h4);
      h4 q1 = __builtin_convertvector(*(const f4*)(ep +  4), h4);
      h4 q2 = __builtin_convertvector(*(const f4*)(ep + 16), h4);
      h4 q3 = __builtin_convertvector(*(const f4*)(ep + 20), h4);
      v8h lo = __builtin_shufflevector(q0, q1, 0,1,2,3,4,5,6,7);
      v8h hh = __builtin_shufflevector(q2, q3, 0,1,2,3,4,5,6,7);
      a = cat8(lo, hh);
    }
#pragma unroll
    for (int g = 0; g < 2; ++g) {  // 4-frag B groups from LDS
      v16h bf[4];
#pragma unroll
      for (int i = 0; i < 4; ++i) {
        const v8h* bp = (const v8h*)(Wl + (size_t)((kt * 8 + g * 4 + i) * 32 + lane) * 16);
        bf[i] = cat8(bp[0], bp[1]);
      }
#pragma unroll
      for (int i = 0; i < 4; ++i)
        acc[g * 4 + i] = WMMA_F16(a, bf[i], acc[g * 4 + i]);
    }
  }
  __syncthreads();

  // D-layout (m = r + 8*hi, n = nt*16 + ml) -> LDS row-major f16
#pragma unroll
  for (int nt = 0; nt < 8; ++nt)
#pragma unroll
    for (int r = 0; r < 8; ++r)
      Dt[(m0 + r + 8 * hi) * LDO16 + nt * 16 + ml] = (_Float16)acc[nt][r];
  __syncthreads();

  {  // coalesced b128 writeout
    int er = t >> 1, kh = t & 1;
    int e = base + er;
    if (e < E) {
      const v8h* src = (const v8h*)&Dt[er * LDO16 + kh * 64];
      v8h* dst = (v8h*)(g1 + (size_t)e * 128 + kh * 64);
#pragma unroll
      for (int j = 0; j < 8; ++j) dst[j] = src[j];
    }
  }
}

// ---------------------------------------------------------------------------
// K2: segmented sum / sumsq of gated. seg is sorted -> few atomic flushes.
// ---------------------------------------------------------------------------
__global__ __launch_bounds__(128) void k_red1(
    const _Float16* __restrict__ g1, const int* __restrict__ seg,
    float* __restrict__ sum1, float* __restrict__ sq1, int E) {
  int f = threadIdx.x;
  int e0 = blockIdx.x * 2048;
  if (e0 >= E) return;
  int e1 = e0 + 2048; if (e1 > E) e1 = E;
  float s = 0.f, q = 0.f;
  int cur = seg[e0];
  for (int e = e0; e < e1; ++e) {
    __builtin_prefetch(&g1[(size_t)(e + 16) * 128 + f], 0, 0);
    int c = seg[e];
    if (c != cur) {
      atomicAdd(&sum1[cur * 128 + f], s);
      atomicAdd(&sq1[cur * 128 + f], q);
      s = 0.f; q = 0.f; cur = c;
    }
    float v = (float)g1[(size_t)e * 128 + f];
    s += v; q += v * v;
  }
  atomicAdd(&sum1[cur * 128 + f], s);
  atomicAdd(&sq1[cur * 128 + f], q);
}

// ---------------------------------------------------------------------------
// K3/K5: finalize mean / rstd per (crystal, feature); count via bsearch.
// ---------------------------------------------------------------------------
__global__ __launch_bounds__(256) void k_fin(
    const float* __restrict__ sum, const float* __restrict__ sq,
    const int* __restrict__ seg, int E, int F,
    float* __restrict__ meanv, float* __restrict__ rstd) {
  int gid = blockIdx.x * blockDim.x + threadIdx.x;
  if (gid >= 256 * F) return;
  int c = gid / F;
  int lo = lower_bound_dev(seg, E, c);
  int hi = lower_bound_dev(seg, E, c + 1);
  float cnt = fmaxf((float)(hi - lo), 1.f);
  float m = sum[gid] / cnt;
  float var = sq[gid] / cnt - m * m;
  meanv[gid] = m;
  rstd[gid] = rsqrtf(var + 1e-5f);
}

// ---------------------------------------------------------------------------
// K4: norm1 -> tanh(filter . W_mask) via wave32 xor-reduce -> relu gate ->
//     stats2 + store S (f16). Lanes 0-15: core feats, lanes 16-31: filter.
// ---------------------------------------------------------------------------
__global__ __launch_bounds__(256) void k_mid(
    const _Float16* __restrict__ g1, const int* __restrict__ seg,
    const float* __restrict__ mean1, const float* __restrict__ rstd1,
    const float* __restrict__ gamma1, const float* __restrict__ beta1,
    const float* __restrict__ wmask, _Float16* __restrict__ S,
    float* __restrict__ sum2, float* __restrict__ sq2, int E) {
  const int lane = threadIdx.x & 31, wid = threadIdx.x >> 5;
  const int CH = 256;
  int e0 = (blockIdx.x * 8 + wid) * CH;
  if (e0 >= E) return;
  int e1 = e0 + CH; if (e1 > E) e1 = E;
  const int f0 = lane * 4;
  const bool isCore = (f0 < 64);

  f4 gm = *(const f4*)(gamma1 + f0);
  f4 bt = *(const f4*)(beta1 + f0);
  f4 wm; wm[0] = wm[1] = wm[2] = wm[3] = 0.f;
  if (!isCore) wm = *(const f4*)(wmask + (f0 - 64));

  float as0 = 0.f, as1 = 0.f, as2 = 0.f, as3 = 0.f;
  float aq0 = 0.f, aq1 = 0.f, aq2 = 0.f, aq3 = 0.f;
  int cur = seg[e0];

  for (int e = e0; e < e1; ++e) {
    __builtin_prefetch(&g1[(size_t)(e + 8) * 128 + f0], 0, 0);
    int c = seg[e];
    if (c != cur) {
      if (isCore) {
        atomicAdd(&sum2[cur * 64 + f0 + 0], as0); atomicAdd(&sq2[cur * 64 + f0 + 0], aq0);
        atomicAdd(&sum2[cur * 64 + f0 + 1], as1); atomicAdd(&sq2[cur * 64 + f0 + 1], aq1);
        atomicAdd(&sum2[cur * 64 + f0 + 2], as2); atomicAdd(&sq2[cur * 64 + f0 + 2], aq2);
        atomicAdd(&sum2[cur * 64 + f0 + 3], as3); atomicAdd(&sq2[cur * 64 + f0 + 3], aq3);
      }
      as0 = as1 = as2 = as3 = 0.f; aq0 = aq1 = aq2 = aq3 = 0.f;
      cur = c;
    }
    h4 gv = *(const h4*)(g1 + (size_t)e * 128 + f0);
    f4 mv = *(const f4*)(mean1 + cur * 128 + f0);
    f4 rv = *(const f4*)(rstd1 + cur * 128 + f0);
    f4 x = (__builtin_convertvector(gv, f4) - mv) * rv * gm + bt;

    float part = isCore ? 0.f : (x[0] * wm[0] + x[1] * wm[1] + x[2] * wm[2] + x[3] * wm[3]);
#pragma unroll
    for (int off = 16; off > 0; off >>= 1) part += __shfl_xor(part, off, 32);
    float filt = tanhf(part);

    if (isCore) {
      float s0 = filt * fmaxf(x[0], 0.f), s1 = filt * fmaxf(x[1], 0.f);
      float s2 = filt * fmaxf(x[2], 0.f), s3 = filt * fmaxf(x[3], 0.f);
      h4 outv; outv[0] = (_Float16)s0; outv[1] = (_Float16)s1;
      outv[2] = (_Float16)s2; outv[3] = (_Float16)s3;
      *(h4*)(S + (size_t)e * 64 + f0) = outv;
      as0 += s0; aq0 += s0 * s0; as1 += s1; aq1 += s1 * s1;
      as2 += s2; aq2 += s2 * s2; as3 += s3; aq3 += s3 * s3;
    }
  }
  if (isCore) {
    atomicAdd(&sum2[cur * 64 + f0 + 0], as0); atomicAdd(&sq2[cur * 64 + f0 + 0], aq0);
    atomicAdd(&sum2[cur * 64 + f0 + 1], as1); atomicAdd(&sq2[cur * 64 + f0 + 1], aq1);
    atomicAdd(&sum2[cur * 64 + f0 + 2], as2); atomicAdd(&sq2[cur * 64 + f0 + 2], aq2);
    atomicAdd(&sum2[cur * 64 + f0 + 3], as3); atomicAdd(&sq2[cur * 64 + f0 + 3], aq3);
  }
}

// ---------------------------------------------------------------------------
// K6: S tile via TDM (padded stride == LDX) -> normalize in place ->
//     2x residual MLP (WMMA, bias/residual via C) -> inv_sqrt2*relu(edge+x).
// ---------------------------------------------------------------------------
#define LDX 72   // halves (TDM: 32 dw + 4 dw pad = 36 dw = 72 halves)
#define LDH 40   // halves
#define LDO 68   // floats

__global__ __launch_bounds__(256) void k_tail(
    const _Float16* __restrict__ S, const int* __restrict__ seg,
    const float* __restrict__ mean2, const float* __restrict__ rstd2,
    const float* __restrict__ gamma2, const float* __restrict__ beta2,
    const _Float16* __restrict__ w1frag, const float* __restrict__ rB1,
    const _Float16* __restrict__ w2frag, const float* __restrict__ rB2,
    const float* __restrict__ edge, float* __restrict__ out, int E) {
  __shared__ __align__(16) _Float16 Xt[128 * LDX];
  __shared__ __align__(16) _Float16 Ht[128 * LDH];
  __shared__ __align__(16) float    Ot[128 * LDO];
  const int t = threadIdx.x;
  const int base = blockIdx.x * 128;
  int rows = E - base; if (rows > 128) rows = 128;

  if (t < 32) {  // wave 0: DMA raw S tile into Xt (OOB rows zero-filled)
    tdm_load_tile_to_lds(S + (size_t)base * 64, (unsigned)(size_t)(void*)Xt,
                         rows, 64, /*2B*/1, /*32dw*/4, /*+4dw*/3);
    __builtin_amdgcn_s_wait_tensorcnt((short)0);
  }
  __syncthreads();

  {  // normalize in place (each thread owns a disjoint 32-half chunk)
    int er = t >> 1, kh = t & 1;
    int e = base + er;
    _Float16* xp = &Xt[er * LDX + kh * 32];
    if (e < E) {
      int c = seg[e];
      const f4* mp = (const f4*)(mean2 + c * 64 + kh * 32);
      const f4* rp = (const f4*)(rstd2 + c * 64 + kh * 32);
      const f4* gp = (const f4*)(gamma2 + kh * 32);
      const f4* bp = (const f4*)(beta2 + kh * 32);
#pragma unroll
      for (int j = 0; j < 8; ++j) {
        f4 v = __builtin_convertvector(*(const h4*)(xp + j * 4), f4);
        f4 x = (v - mp[j]) * rp[j] * gp[j] + bp[j];
        *(h4*)(xp + j * 4) = __builtin_convertvector(x, h4);
      }
    }
  }
  __syncthreads();

  const int wid = t >> 5, lane = t & 31;
  const int ml = lane & 15, hi = lane >> 4;
  const int m0 = wid * 16;  // each wave touches only rows [m0, m0+16)

#pragma unroll
  for (int layer = 0; layer < 2; ++layer) {
    // h = relu(x @ rW1 + rB1): K=64 (2 steps), N=32 (2 tiles)
    v8f hacc[2];
#pragma unroll
    for (int nt = 0; nt < 2; ++nt) {
      float bv = rB1[layer * 32 + nt * 16 + ml];
#pragma unroll
      for (int r = 0; r < 8; ++r) hacc[nt][r] = bv;
    }
#pragma unroll
    for (int kt = 0; kt < 2; ++kt) {
      const _Float16* ap = &Xt[(m0 + ml) * LDX + kt * 32 + hi * 8];
      v16h a = cat8(*(const v8h*)ap, *(const v8h*)(ap + 16));
#pragma unroll
      for (int nt = 0; nt < 2; ++nt) {
        v16h b = *(const v16h*)(w1frag + (size_t)((((layer * 2 + kt) * 2 + nt) * 32 + lane)) * 16);
        hacc[nt] = WMMA_F16(a, b, hacc[nt]);
      }
    }
    // relu(h) -> Ht (wave-private rows, in-wave DS ordering)
#pragma unroll
    for (int nt = 0; nt < 2; ++nt)
#pragma unroll
      for (int r = 0; r < 8; ++r)
        Ht[(m0 + r + 8 * hi) * LDH + nt * 16 + ml] = (_Float16)fmaxf(hacc[nt][r], 0.f);

    // x = x + h @ rW2 + rB2: K=32 (1 step), N=64 (4 tiles); residual via C.
    const _Float16* ap2 = &Ht[(m0 + ml) * LDH + hi * 8];
    v16h a2 = cat8(*(const v8h*)ap2, *(const v8h*)(ap2 + 16));
    v8f xacc[4];
#pragma unroll
    for (int nt = 0; nt < 4; ++nt) {
      float bv = rB2[layer * 64 + nt * 16 + ml];
#pragma unroll
      for (int r = 0; r < 8; ++r)
        xacc[nt][r] = (float)Xt[(m0 + r + 8 * hi) * LDX + nt * 16 + ml] + bv;
      v16h b = *(const v16h*)(w2frag + (size_t)(((layer * 4 + nt) * 32 + lane)) * 16);
      xacc[nt] = WMMA_F16(a2, b, xacc[nt]);
    }
    if (layer == 0) {
#pragma unroll
      for (int nt = 0; nt < 4; ++nt)
#pragma unroll
        for (int r = 0; r < 8; ++r)
          Xt[(m0 + r + 8 * hi) * LDX + nt * 16 + ml] = (_Float16)xacc[nt][r];
    } else {
#pragma unroll
      for (int nt = 0; nt < 4; ++nt)
#pragma unroll
        for (int r = 0; r < 8; ++r)
          Ot[(m0 + r + 8 * hi) * LDO + nt * 16 + ml] = xacc[nt][r];
    }
  }
  __syncthreads();

  {  // out = inv_sqrt2 * relu(edge + x), coalesced
    int er = t >> 1, kh = t & 1;
    int e = base + er;
    if (e < E) {
      const f4* ep = (const f4*)(edge + (size_t)e * 64 + kh * 32);
      f4* op = (f4*)(out + (size_t)e * 64 + kh * 32);
      const f4* xs = (const f4*)&Ot[er * LDO + kh * 32];
#pragma unroll
      for (int j = 0; j < 8; ++j) {
        f4 r = ep[j] + xs[j];
        f4 o;
        o[0] = 0.70710678118654752f * fmaxf(r[0], 0.f);
        o[1] = 0.70710678118654752f * fmaxf(r[1], 0.f);
        o[2] = 0.70710678118654752f * fmaxf(r[2], 0.f);
        o[3] = 0.70710678118654752f * fmaxf(r[3], 0.f);
        op[j] = o;
      }
    }
  }
}

// ---------------------------------------------------------------------------
// Host launcher
// ---------------------------------------------------------------------------
extern "C" void kernel_launch(void* const* d_in, const int* in_sizes, int n_in,
                              void* d_out, int out_size, void* d_ws, size_t ws_size,
                              hipStream_t stream) {
  (void)n_in; (void)out_size; (void)ws_size;
  const float* atom   = (const float*)d_in[0];
  const float* edge   = (const float*)d_in[1];
  const int*   nidx   = (const int*)d_in[2];
  const int*   seg    = (const int*)d_in[3];
  const float* Wfull  = (const float*)d_in[4];
  const float* Wmask  = (const float*)d_in[5];
  const float* gamma1 = (const float*)d_in[6];
  const float* beta1  = (const float*)d_in[7];
  const float* gamma2 = (const float*)d_in[8];
  const float* beta2  = (const float*)d_in[9];
  const float* rW1    = (const float*)d_in[10];
  const float* rB1    = (const float*)d_in[11];
  const float* rW2    = (const float*)d_in[12];
  const float* rB2    = (const float*)d_in[13];
  const int E = in_sizes[1] / 64;

  char* ws = (char*)d_ws;
  _Float16* wfull_f = (_Float16*)(ws + 0);        // 32768 B
  _Float16* w1_f    = (_Float16*)(ws + 32768);    // 8192 B
  _Float16* w2_f    = (_Float16*)(ws + 40960);    // 8192 B
  float* sum1  = (float*)(ws + 49152);            // 131072 B
  float* sq1   = (float*)(ws + 180224);           // 131072 B
  float* mean1 = (float*)(ws + 311296);           // 131072 B
  float* rstd1 = (float*)(ws + 442368);           // 131072 B
  float* sum2  = (float*)(ws + 573440);           // 65536 B
  float* sq2   = (float*)(ws + 638976);           // 65536 B
  float* mean2 = (float*)(ws + 704512);           // 65536 B
  float* rstd2 = (float*)(ws + 770048);           // 65536 B
  _Float16* G1 = (_Float16*)(ws + 835584);                            // E*128*2 B
  _Float16* Sb = (_Float16*)(ws + 835584 + (size_t)E * 128 * 2);      // E*64*2 B

  const int tiles = (E + 127) / 128;
  const int chunks = (E + 2047) / 2048;

  k_prep<<<128, 256, 0, stream>>>(Wfull, rW1, rW2, wfull_f, w1_f, w2_f,
                                  sum1, sq1, sum2, sq2);
  k_gemm1<<<tiles, 256, 0, stream>>>(atom, edge, nidx, wfull_f, G1, E);
  k_red1<<<chunks, 128, 0, stream>>>(G1, seg, sum1, sq1, E);
  k_fin<<<(256 * 128 + 255) / 256, 256, 0, stream>>>(sum1, sq1, seg, E, 128, mean1, rstd1);
  k_mid<<<chunks, 256, 0, stream>>>(G1, seg, mean1, rstd1, gamma1, beta1,
                                    Wmask, Sb, sum2, sq2, E);
  k_fin<<<(256 * 64 + 255) / 256, 256, 0, stream>>>(sum2, sq2, seg, E, 64, mean2, rstd2);
  k_tail<<<tiles, 256, 0, stream>>>(Sb, seg, mean2, rstd2, gamma2, beta2,
                                    w1_f, rB1, w2_f, rB2, edge, (float*)d_out, E);
}
